// WindowAttention_29368986370523
// MI455X (gfx1250) — matrix operand order repaired and checked by reading, not compiled, not run on hardware
//
#include <hip/hip_runtime.h>

// ---------------------------------------------------------------------------
// CDNA5 (gfx1250) fused Swin window-attention, bf16 WMMA pipeline.
//   DIM=384, WIN=7 (N=49), HEADS=12, HEAD_DIM=32, B=2048 windows.
// Pipeline:
//   0) convert x / qkv_w / proj_w -> bf16; build padded bias/mask fp32 tables
//   1) QKV GEMM   [100352,384]x[1152,384]^T  (async double-buffered WMMA)
//   2) attention  per (window, head): WMMA qk^T -> softmax -> WMMA attn*v
//   3) proj GEMM  [100352,384]x[384,384]^T -> fp32 out (+bias)
// ---------------------------------------------------------------------------

typedef __attribute__((ext_vector_type(16))) __bf16 v16bf;
typedef __attribute__((ext_vector_type(8)))  float  v8f;
typedef int v4i __attribute__((vector_size(4 * sizeof(int))));

#define WMMA_BF16(a, b, c) \
    __builtin_amdgcn_wmma_f32_16x16x32_bf16(false, (a), false, (b), (short)0, (c), false, false)

union BF16Frag {
    v16bf v;
    unsigned short u[16];
};

// fp32 -> bf16 round-to-nearest-even (integer path)
__device__ __forceinline__ unsigned short f2bf(float f) {
    unsigned int x = __float_as_uint(f);
    x += 0x7FFFu + ((x >> 16) & 1u);
    return (unsigned short)(x >> 16);
}

// ---------------------------------------------------------------------------
// Async global->LDS staging (CDNA5). Guarded: falls back to sync loads.
// Builtin signature (from hipcc diagnostic): (v4i AS1*, v4i AS3*, imm, imm)
// ---------------------------------------------------------------------------
#if defined(__has_builtin)
#if __has_builtin(__builtin_amdgcn_global_load_async_to_lds_b128)
#define HAVE_ASYNC_LDS 1
#endif
#endif
#ifndef HAVE_ASYNC_LDS
#define HAVE_ASYNC_LDS 0
#endif

#if HAVE_ASYNC_LDS
__device__ __forceinline__ void async_g2l_b128(const unsigned short* g, unsigned short* l) {
    __builtin_amdgcn_global_load_async_to_lds_b128(
        (__attribute__((address_space(1))) v4i*)g,
        (__attribute__((address_space(3))) v4i*)l,
        0, 0);
}
#define WAIT_ASYNC_4() asm volatile("s_wait_asynccnt 4" ::: "memory")
#define WAIT_ASYNC_0() asm volatile("s_wait_asynccnt 0" ::: "memory")
#endif

// ---------------------------------------------------------------------------
// Fragment builders (LDS -> VGPR), ISA 7.12.2 wave32 layouts.
//   A 16x32 bf16:  VGPR j<4: K = half*8+2j ; j>=4: K = 16+half*8+2(j-4)
//   B 32x16 bf16 ([n][k] storage, row n contiguous over k): same K as A
// ---------------------------------------------------------------------------
__device__ __forceinline__ v16bf load_a_frag(const unsigned short* tile,
                                             int row0, int stride, int k0, int lane) {
    const int half = lane >> 4, r = lane & 15;
    const unsigned short* rp = tile + (row0 + r) * stride + k0;
    BF16Frag f;
#pragma unroll
    for (int j = 0; j < 8; ++j) {
        const int k = (j < 4) ? (half * 8 + 2 * j) : (16 + half * 8 + 2 * (j - 4));
        f.u[2 * j]     = rp[k];
        f.u[2 * j + 1] = rp[k + 1];
    }
    return f.v;
}

__device__ __forceinline__ v16bf load_b_frag_nk(const unsigned short* tile,
                                                int n0, int stride, int k0, int lane) {
    const int half = lane >> 4, r = lane & 15;
    const unsigned short* rp = tile + (n0 + r) * stride + k0;
    BF16Frag f;
#pragma unroll
    for (int j = 0; j < 8; ++j) {
        const int k = (j < 4) ? (half * 8 + 2 * j) : (16 + half * 8 + 2 * (j - 4));
        f.u[2 * j]     = rp[k];
        f.u[2 * j + 1] = rp[k + 1];
    }
    return f.v;
}

// ---------------------------------------------------------------------------
// Kernel 0a: fp32 -> bf16 conversion (x and weights)
// ---------------------------------------------------------------------------
__global__ __launch_bounds__(256) void k_convert_bf16(const float* __restrict__ in,
                                                      unsigned short* __restrict__ out,
                                                      int n) {
    int i = blockIdx.x * 256 + threadIdx.x;
    if (i < n) out[i] = f2bf(in[i]);
}

// ---------------------------------------------------------------------------
// Kernel 0b: padded bias table  bias64[h][64][64] (pad = -1e30)
// ---------------------------------------------------------------------------
__global__ __launch_bounds__(256) void k_make_bias64(const float* __restrict__ rpb,
                                                     const int* __restrict__ rel,
                                                     float* __restrict__ bias64) {
    int idx = blockIdx.x * 256 + threadIdx.x;   // 12*4096
    if (idx >= 12 * 4096) return;
    const int h = idx >> 12, m = (idx >> 6) & 63, n = idx & 63;
    float v = -1e30f;
    if (m < 49 && n < 49) v = rpb[rel[m * 49 + n] * 12 + h];
    bias64[idx] = v;
}

// ---------------------------------------------------------------------------
// Kernel 0c: padded mask table  mask64[w][64][64] (masked = -1e30, else 0)
// ---------------------------------------------------------------------------
__global__ __launch_bounds__(256) void k_make_mask64(const unsigned char* __restrict__ mask,
                                                     float* __restrict__ mask64) {
    int idx = blockIdx.x * 256 + threadIdx.x;   // 64*4096
    if (idx >= 64 * 4096) return;
    const int w = idx >> 12, m = (idx >> 6) & 63, n = idx & 63;
    float v = 0.f;
    if (m < 49 && n < 49 && mask[w * 2401 + m * 49 + n]) v = -1e30f;
    mask64[idx] = v;
}

// ---------------------------------------------------------------------------
// Tiled GEMM: C[M,N] = A[M,K]*W[N,K]^T + bias. A,W bf16; fp32 accum.
//   block 256 = 8 waves (2x4), tile 128x128, K-step 32, double-buffered LDS,
//   async global->LDS staging one stage ahead.
// ---------------------------------------------------------------------------
template <bool OUT_IS_BF16>
__global__ __launch_bounds__(256) void k_gemm_wmma(const unsigned short* __restrict__ A,
                                                   const unsigned short* __restrict__ W,
                                                   const float* __restrict__ bias,
                                                   void* __restrict__ Cptr,
                                                   int K, int Nout) {
    constexpr int LS = 40;                       // row stride (ushorts); 80B = 16B aligned
    __shared__ unsigned short As[2][128 * LS];
    __shared__ unsigned short Bs[2][128 * LS];

    const int m0   = blockIdx.y * 128;
    const int n0   = blockIdx.x * 128;
    const int tid  = threadIdx.x;
    const int lane = tid & 31;
    const int wv   = tid >> 5;
    const int wm   = wv >> 2;                    // 0..1 -> 64-row band
    const int wn   = wv & 3;                     // 0..3 -> 32-col band

    v8f acc[4][2];
#pragma unroll
    for (int mi = 0; mi < 4; ++mi)
#pragma unroll
        for (int ni = 0; ni < 2; ++ni) acc[mi][ni] = v8f{};

    const int nk = K >> 5;

#if HAVE_ASYNC_LDS
    auto stage = [&](int kk, int buf) {
#pragma unroll
        for (int i = 0; i < 2; ++i) {            // 512 slots of 16B per tile
            const int slot = tid + i * 256;
            const int row  = slot >> 2, c8 = (slot & 3) * 8;
            async_g2l_b128(A + (size_t)(m0 + row) * K + kk + c8, &As[buf][row * LS + c8]);
            async_g2l_b128(W + (size_t)(n0 + row) * K + kk + c8, &Bs[buf][row * LS + c8]);
        }
    };
    stage(0, 0);
    for (int s = 0; s < nk; ++s) {
        const int cur = s & 1;
        if (s + 1 < nk) {
            stage((s + 1) << 5, cur ^ 1);
            WAIT_ASYNC_4();                      // oldest 4 (current stage) done
        } else {
            WAIT_ASYNC_0();
        }
        __syncthreads();
        const v16bf b0 = load_b_frag_nk(Bs[cur], wn * 32 + 0,  LS, 0, lane);
        const v16bf b1 = load_b_frag_nk(Bs[cur], wn * 32 + 16, LS, 0, lane);
#pragma unroll
        for (int mi = 0; mi < 4; ++mi) {
            const v16bf a = load_a_frag(As[cur], wm * 64 + mi * 16, LS, 0, lane);
            acc[mi][0] = WMMA_BF16(a, b0, acc[mi][0]);
            acc[mi][1] = WMMA_BF16(a, b1, acc[mi][1]);
        }
        __syncthreads();                         // protect buf before reuse
    }
#else
    for (int s = 0; s < nk; ++s) {
        const int kk = s << 5;
#pragma unroll
        for (int i = 0; i < 2; ++i) {
            const int slot = tid + i * 256;
            const int row  = slot >> 2, c8 = (slot & 3) * 8;
            *(uint4*)&As[0][row * LS + c8] = *(const uint4*)(A + (size_t)(m0 + row) * K + kk + c8);
            *(uint4*)&Bs[0][row * LS + c8] = *(const uint4*)(W + (size_t)(n0 + row) * K + kk + c8);
        }
        __syncthreads();
        const v16bf b0 = load_b_frag_nk(Bs[0], wn * 32 + 0,  LS, 0, lane);
        const v16bf b1 = load_b_frag_nk(Bs[0], wn * 32 + 16, LS, 0, lane);
#pragma unroll
        for (int mi = 0; mi < 4; ++mi) {
            const v16bf a = load_a_frag(As[0], wm * 64 + mi * 16, LS, 0, lane);
            acc[mi][0] = WMMA_BF16(a, b0, acc[mi][0]);
            acc[mi][1] = WMMA_BF16(a, b1, acc[mi][1]);
        }
        __syncthreads();
    }
#endif

    // ---- epilogue: bias + store ----
    const int half = lane >> 4, r = lane & 15;
#pragma unroll
    for (int mi = 0; mi < 4; ++mi) {
#pragma unroll
        for (int ni = 0; ni < 2; ++ni) {
            const int n  = n0 + wn * 32 + ni * 16 + r;
            const float bv = bias[n];
#pragma unroll
            for (int i = 0; i < 8; ++i) {
                const int m = m0 + wm * 64 + mi * 16 + i + half * 8;
                const float val = acc[mi][ni][i] + bv;
                if constexpr (OUT_IS_BF16)
                    ((unsigned short*)Cptr)[(size_t)m * Nout + n] = f2bf(val);
                else
                    ((float*)Cptr)[(size_t)m * Nout + n] = val;
            }
        }
    }
}

// ---------------------------------------------------------------------------
// Fused attention: one block per (window b, head h). 128 threads = 4 waves.
//   Branch-free epilogue via precomputed bias64/mask64 tables.
// ---------------------------------------------------------------------------
__global__ __launch_bounds__(128) void k_attention(const unsigned short* __restrict__ qkv,
                                                   const float* __restrict__ bias64,
                                                   const float* __restrict__ mask64,
                                                   unsigned short* __restrict__ out) {
    constexpr int QS = 36;   // LDS stride for q/k [64][32]
    constexpr int VT = 72;   // LDS stride for v^T [32][64]
    constexpr int AS = 68;   // LDS stride for attn [64][64]
    __shared__ unsigned short q_s[64 * QS];
    __shared__ unsigned short k_s[64 * QS];
    __shared__ unsigned short v_t[32 * VT];      // transposed: v_t[d][token]
    __shared__ float          att[64 * AS];
    __shared__ unsigned short attb[64 * AS];

    const int b    = blockIdx.x / 12;
    const int h    = blockIdx.x % 12;
    const int tid  = threadIdx.x;
    const int lane = tid & 31;
    const int wv   = tid >> 5;                   // wave id == M-tile index
    const int half = lane >> 4, r = lane & 15;

    // ---- stage q, k (row-major) and v (transposed); zero-pad rows 49..63 ----
    for (int slot = tid; slot < 512; slot += 128) {       // 64 rows x 8 chunks of 4
        const int row = slot >> 3, c4 = (slot & 7) * 4;
        uint2 qv = {0u, 0u}, kv = {0u, 0u}, vv = {0u, 0u};
        if (row < 49) {
            const size_t base = ((size_t)(b * 49 + row)) * 1152 + h * 32 + c4;
            qv = *(const uint2*)(qkv + base);
            kv = *(const uint2*)(qkv + base + 384);
            vv = *(const uint2*)(qkv + base + 768);
        }
        *(uint2*)&q_s[row * QS + c4] = qv;
        *(uint2*)&k_s[row * QS + c4] = kv;
        v_t[(c4 + 0) * VT + row] = (unsigned short)(vv.x & 0xFFFFu);
        v_t[(c4 + 1) * VT + row] = (unsigned short)(vv.x >> 16);
        v_t[(c4 + 2) * VT + row] = (unsigned short)(vv.y & 0xFFFFu);
        v_t[(c4 + 3) * VT + row] = (unsigned short)(vv.y >> 16);
    }
    __syncthreads();

    // ---- attn = q k^T * scale + bias64 + mask64 (branch-free) ----
    const float scale = 0.17677669529663687f;    // 32^-0.5
    const float* bh = bias64 + h * 4096;
    const float* mw = mask64 + (size_t)(b & 63) * 4096;
    {
        const int mi = wv;
        const v16bf a = load_a_frag(q_s, mi * 16, QS, 0, lane);
#pragma unroll
        for (int ni = 0; ni < 4; ++ni) {
            const v16bf bf = load_b_frag_nk(k_s, ni * 16, QS, 0, lane);
            v8f c = v8f{};
            c = WMMA_BF16(a, bf, c);
            const int n = ni * 16 + r;
#pragma unroll
            for (int i = 0; i < 8; ++i) {
                const int m = mi * 16 + i + half * 8;
                att[m * AS + n] = c[i] * scale + bh[m * 64 + n] + mw[m * 64 + n];
            }
        }
    }
    __syncthreads();

    // ---- row softmax (fp32), requantize to bf16 ----
    if (tid < 49) {
        float mx = -1e30f;
        for (int j = 0; j < 49; ++j) mx = fmaxf(mx, att[tid * AS + j]);
        float s = 0.f;
        for (int j = 0; j < 49; ++j) {
            const float e = __expf(att[tid * AS + j] - mx);
            att[tid * AS + j] = e;
            s += e;
        }
        const float inv = 1.0f / s;
        for (int j = 0; j < 64; ++j)
            attb[tid * AS + j] = (j < 49) ? f2bf(att[tid * AS + j] * inv) : 0;
    } else if (tid < 64) {
        for (int j = 0; j < 64; ++j) attb[tid * AS + j] = 0;
    }
    __syncthreads();

    // ---- out = attn @ v : M tiles 4 (one per wave), N tiles 2, K = 64 ----
    {
        const int mi = wv;
        v8f c0 = v8f{}, c1 = v8f{};
#pragma unroll
        for (int ks = 0; ks < 2; ++ks) {
            const v16bf a  = load_a_frag(attb, mi * 16, AS, ks * 32, lane);
            const v16bf b0 = load_b_frag_nk(v_t, 0,  VT, ks * 32, lane);   // rows d=0..15
            const v16bf b1 = load_b_frag_nk(v_t, 16, VT, ks * 32, lane);   // rows d=16..31
            c0 = WMMA_BF16(a, b0, c0);
            c1 = WMMA_BF16(a, b1, c1);
        }
#pragma unroll
        for (int i = 0; i < 8; ++i) {
            const int m = mi * 16 + i + half * 8;
            if (m < 49) {
                const size_t base = ((size_t)(b * 49 + m)) * 384 + h * 32;
                out[base + r]      = f2bf(c0[i]);
                out[base + 16 + r] = f2bf(c1[i]);
            }
        }
    }
}

// ---------------------------------------------------------------------------
// Host-side orchestration
// ---------------------------------------------------------------------------
extern "C" void kernel_launch(void* const* d_in, const int* in_sizes, int n_in,
                              void* d_out, int out_size, void* d_ws, size_t ws_size,
                              hipStream_t stream) {
    const float*         x       = (const float*)d_in[0];
    const unsigned char* mask    = (const unsigned char*)d_in[1];   // bool
    const float*         qkv_w   = (const float*)d_in[2];
    const float*         qkv_b   = (const float*)d_in[3];
    const float*         proj_w  = (const float*)d_in[4];
    const float*         proj_b  = (const float*)d_in[5];
    const float*         rpb     = (const float*)d_in[6];
    const int*           rel_idx = (const int*)d_in[7];

    const int B   = 2048;
    const int N   = 49;
    const int DIM = 384;
    const int M   = B * N;          // 100352
    const int NQ  = 3 * DIM;        // 1152

    // workspace layout (all offsets 16B aligned)
    char* ws = (char*)d_ws;
    unsigned short* qkvw_bf  = (unsigned short*)(ws + 0);            // 442368 bf16
    unsigned short* projw_bf = (unsigned short*)(ws + 884736);       // 147456 bf16
    float*          bias64   = (float*)(ws + 1179648);               // 12*4096 f32
    float*          mask64   = (float*)(ws + 1376256);               // 64*4096 f32
    unsigned short* x_bf     = (unsigned short*)(ws + 2424832);      // M*384 bf16
    unsigned short* qkv_act  = (unsigned short*)(ws + 79495168ull);  // M*1152 bf16
    unsigned short* attn_out = (unsigned short*)(ws + 310706176ull); // M*384 bf16

    // 0) conversions + tables
    {
        const int nx = M * DIM;        // 38535168
        const int n1 = NQ * DIM;       // 442368
        const int n2 = DIM * DIM;      // 147456
        k_convert_bf16<<<(nx + 255) / 256, 256, 0, stream>>>(x, x_bf, nx);
        k_convert_bf16<<<(n1 + 255) / 256, 256, 0, stream>>>(qkv_w, qkvw_bf, n1);
        k_convert_bf16<<<(n2 + 255) / 256, 256, 0, stream>>>(proj_w, projw_bf, n2);
        k_make_bias64<<<(12 * 4096 + 255) / 256, 256, 0, stream>>>(rpb, rel_idx, bias64);
        k_make_mask64<<<(64 * 4096 + 255) / 256, 256, 0, stream>>>(mask, mask64);
    }

    // 1) QKV projection: bf16 [M,384] x [1152,384]^T -> bf16 [M,1152]
    k_gemm_wmma<true><<<dim3(NQ / 128, M / 128), 256, 0, stream>>>(
        x_bf, qkvw_bf, qkv_b, (void*)qkv_act, DIM, NQ);

    // 2) fused window attention per (window, head) -> bf16 [M,384]
    k_attention<<<B * 12, 128, 0, stream>>>(qkv_act, bias64, mask64, attn_out);

    // 3) output projection: bf16 [M,384] x [384,384]^T -> fp32 d_out (+bias)
    k_gemm_wmma<false><<<dim3(DIM / 128, M / 128), 256, 0, stream>>>(
        attn_out, projw_bf, proj_b, (void*)d_out, DIM, DIM);

    (void)in_sizes; (void)n_in; (void)out_size; (void)ws_size;
}
